// Match_38457137168646
// MI455X (gfx1250) — matrix operand order repaired and checked by reading, not compiled, not run on hardware
//
#include <hip/hip_runtime.h>

// ---------------------------------------------------------------------------
// CDNA5 / gfx1250 implementation. wave32, WMMA 16x16x32 bf16 (f32 accum),
// async global->LDS staging for the shared B tile in the dominant edge GEMM.
// ---------------------------------------------------------------------------

typedef __attribute__((ext_vector_type(16))) __bf16 v16bf;
typedef __attribute__((ext_vector_type(8)))  __bf16 v8bf;
typedef __attribute__((ext_vector_type(8)))  float  v8f;

__device__ __forceinline__ __bf16 f2bf(float x) { return (__bf16)x; }

#define N_NODES 512
#define C_NODE  151
#define C_EDGE  51

// ---------------------------------------------------------------------------
// Zero-fill (h_edge_emb is exactly zeros in the reference).
// ---------------------------------------------------------------------------
__global__ void zero_f4(float4* __restrict__ p, int n4) {
    int i = blockIdx.x * blockDim.x + threadIdx.x;
    if (i < n4) p[i] = make_float4(0.f, 0.f, 0.f, 0.f);
}

// ---------------------------------------------------------------------------
// Build Bt[c][k] = (c < C) ? S[k*C + c] : 0   (bf16, [Cpad][K])
// Matches WMMA B-fragment striping: lane reads 16 contiguous k at fixed col.
// ---------------------------------------------------------------------------
__global__ void prep_bt(const float* __restrict__ S, __bf16* __restrict__ Bt,
                        int K, int C, int Cpad) {
    int idx = blockIdx.x * blockDim.x + threadIdx.x;
    if (idx >= Cpad * K) return;
    int c = idx / K, k = idx - c * K;
    float v = (c < C) ? S[(size_t)k * C + c] : 0.0f;
    Bt[idx] = f2bf(v);
}

// ---------------------------------------------------------------------------
// Build R[n][k] = (k < C) ? S[n*C + k] : 0  (bf16, [D][Cpad]) — schema^T's
// column n as a contiguous K-run, used as B for the h_node GEMM.
// ---------------------------------------------------------------------------
__global__ void prep_rows(const float* __restrict__ S, __bf16* __restrict__ R,
                          int D, int C, int Cpad) {
    int idx = blockIdx.x * blockDim.x + threadIdx.x;
    if (idx >= D * Cpad) return;
    int n = idx / Cpad, k = idx - n * Cpad;
    R[idx] = (k < C) ? f2bf(S[(size_t)n * C + k]) : f2bf(0.0f);
}

// ---------------------------------------------------------------------------
// EDGE GEMM: A fp32 [8192][2048] x Bt bf16 [64][2048] -> fp32 [8192][51].
// One wave per 16-row m-tile, 4 column tiles held in accumulators so each A
// fragment is reused by 4 WMMAs. The 4KB B chunk (shared by all 4 waves of
// the block) is staged to LDS with GLOBAL_LOAD_ASYNC_TO_LDS_B128 (ASYNCcnt).
// ---------------------------------------------------------------------------
__global__ void gemm_edge_wmma(const float* __restrict__ A,
                               const __bf16* __restrict__ Bt,
                               float* __restrict__ out) {
    __shared__ __align__(32) __bf16 sB[64 * 32];   // [col][k-chunk] = 4 KB

    const int tid  = threadIdx.x;                  // 128 threads = 4 waves
    const int wv   = tid >> 5;
    const int lane = tid & 31;
    const int mt   = blockIdx.x * 4 + wv;          // m-tile 0..511
    const int r    = lane & 15;
    const int h    = lane >> 4;

    const float* arow = A + (size_t)(mt * 16 + r) * 2048;

    // This thread's two staging transfers (256 x 16B = 4KB per chunk).
    const int c0 = tid >> 2,        p0 = tid & 3;
    const int c1 = (tid + 128) >> 2, p1 = (tid + 128) & 3;
    const unsigned lds0 = (unsigned)(size_t)&sB[c0 * 32 + p0 * 8];
    const unsigned lds1 = (unsigned)(size_t)&sB[c1 * 32 + p1 * 8];

    v8f acc[4] = {};
    for (int k0 = 0; k0 < 2048; k0 += 32) {
        // Kick off async DMA of the B chunk into LDS (tracked by ASYNCcnt).
        {
            const __bf16* g0 = Bt + c0 * 2048 + k0 + p0 * 8;
            const __bf16* g1 = Bt + c1 * 2048 + k0 + p1 * 8;
            asm volatile("global_load_async_to_lds_b128 %0, %1, off"
                         :: "v"(lds0), "v"(g0) : "memory");
            asm volatile("global_load_async_to_lds_b128 %0, %1, off"
                         :: "v"(lds1), "v"(g1) : "memory");
        }
        // Prefetch A two chunks ahead while DMA is in flight.
        if (k0 + 96 < 2048) __builtin_prefetch(arow + k0 + 64, 0, 1);

        // A fragment: K = {k0+h*16 .. +15} (contiguous, matches B).
        const float4* q1 = (const float4*)(arow + k0 + h * 16);
        const float4* q2 = (const float4*)(arow + k0 + h * 16 + 8);
        float4 a0 = q1[0], a1 = q1[1];
        float4 a2 = q2[0], a3 = q2[1];
        v16bf af;
        af[0]  = f2bf(a0.x); af[1]  = f2bf(a0.y); af[2]  = f2bf(a0.z); af[3]  = f2bf(a0.w);
        af[4]  = f2bf(a1.x); af[5]  = f2bf(a1.y); af[6]  = f2bf(a1.z); af[7]  = f2bf(a1.w);
        af[8]  = f2bf(a2.x); af[9]  = f2bf(a2.y); af[10] = f2bf(a2.z); af[11] = f2bf(a2.w);
        af[12] = f2bf(a3.x); af[13] = f2bf(a3.y); af[14] = f2bf(a3.z); af[15] = f2bf(a3.w);

        // Our own async transfers done -> with the barrier, whole chunk is in.
        asm volatile("s_wait_asynccnt 0x0" ::: "memory");
        __syncthreads();

        #pragma unroll
        for (int ct = 0; ct < 4; ++ct) {
            // B fragment: col = ct*16+r, K pairs = k0 + h*16 .. +15 (contig).
            v16bf bf = *(const v16bf*)&sB[(ct * 16 + r) * 32 + h * 16];
            acc[ct] = __builtin_amdgcn_wmma_f32_16x16x32_bf16(
                false, af, false, bf, (short)0, acc[ct], false, false);
        }
        __syncthreads();   // protect sB before next chunk overwrites it
    }

    #pragma unroll
    for (int ct = 0; ct < 4; ++ct) {
        int col = ct * 16 + r;
        if (col < C_EDGE) {
            #pragma unroll
            for (int v = 0; v < 8; ++v) {
                int row = mt * 16 + h * 8 + v;      // D layout: VGPR v -> M=h*8+v
                out[(size_t)row * C_EDGE + col] = acc[ct][v];
            }
        }
    }
}

// ---------------------------------------------------------------------------
// GEMM: A fp32 [M][K] x Bt bf16 [Cpad][K] -> out fp32 [M][C]  (col<C stored).
// One wave per 16x16 output tile (used for the small node-class GEMM where
// tile count, not data reuse, dominates).
// ---------------------------------------------------------------------------
__global__ void gemm_a32_bt16(const float* __restrict__ A,
                              const __bf16* __restrict__ Bt,
                              float* __restrict__ out,
                              int M, int K, int C, int CT) {
    int wave = blockIdx.x * (blockDim.x >> 5) + (threadIdx.x >> 5);
    int lane = threadIdx.x & 31;
    int mt = wave / CT;
    int ct = wave - mt * CT;
    if (mt * 16 >= M) return;

    int r = lane & 15, h = lane >> 4;
    const float*  arow = A  + (size_t)(mt * 16 + r) * K;
    const __bf16* brow = Bt + (size_t)(ct * 16 + r) * K;

    v8f acc = {};
    for (int k0 = 0; k0 < K; k0 += 32) {
        const float4* p1 = (const float4*)(arow + k0 + h * 16);
        const float4* p2 = (const float4*)(arow + k0 + h * 16 + 8);
        float4 a0 = p1[0], a1 = p1[1];
        float4 a2 = p2[0], a3 = p2[1];
        v16bf af;
        af[0]  = f2bf(a0.x); af[1]  = f2bf(a0.y); af[2]  = f2bf(a0.z); af[3]  = f2bf(a0.w);
        af[4]  = f2bf(a1.x); af[5]  = f2bf(a1.y); af[6]  = f2bf(a1.z); af[7]  = f2bf(a1.w);
        af[8]  = f2bf(a2.x); af[9]  = f2bf(a2.y); af[10] = f2bf(a2.z); af[11] = f2bf(a2.w);
        af[12] = f2bf(a3.x); af[13] = f2bf(a3.y); af[14] = f2bf(a3.z); af[15] = f2bf(a3.w);
        v16bf bf = *(const v16bf*)(brow + k0 + h * 16);
        acc = __builtin_amdgcn_wmma_f32_16x16x32_bf16(
            false, af, false, bf, (short)0, acc, false, false);
    }

    int col = ct * 16 + r;
    if (col < C) {
        #pragma unroll
        for (int v = 0; v < 8; ++v) {
            int row = mt * 16 + h * 8 + v;
            out[(size_t)row * C + col] = acc[v];
        }
    }
}

// ---------------------------------------------------------------------------
// GEMM: A bf16 [M][Kp] x B bf16 [N][Kp] (rows are schema^T cols) -> fp32 [M][N]
// ---------------------------------------------------------------------------
__global__ void gemm_a16_b16(const __bf16* __restrict__ A,
                             const __bf16* __restrict__ B,
                             float* __restrict__ out,
                             int M, int Kp, int N) {
    int NT = N >> 4;
    int wave = blockIdx.x * (blockDim.x >> 5) + (threadIdx.x >> 5);
    int lane = threadIdx.x & 31;
    int mt = wave / NT;
    int nt = wave - mt * NT;
    if (mt * 16 >= M) return;

    int r = lane & 15, h = lane >> 4;
    const __bf16* arow = A + (size_t)(mt * 16 + r) * Kp;
    const __bf16* brow = B + (size_t)(nt * 16 + r) * Kp;

    v8f acc = {};
    for (int k0 = 0; k0 < Kp; k0 += 32) {
        v8bf c1 = *(const v8bf*)(arow + k0 + h * 16);
        v8bf c2 = *(const v8bf*)(arow + k0 + h * 16 + 8);
        v16bf af;
        #pragma unroll
        for (int j = 0; j < 8; ++j) { af[j] = c1[j]; af[8 + j] = c2[j]; }
        v16bf bf = *(const v16bf*)(brow + k0 + h * 16);
        acc = __builtin_amdgcn_wmma_f32_16x16x32_bf16(
            false, af, false, bf, (short)0, acc, false, false);
    }

    int col = nt * 16 + r;
    #pragma unroll
    for (int v = 0; v < 8; ++v) {
        int row = mt * 16 + h * 8 + v;
        out[(size_t)row * N + col] = acc[v];
    }
}

// ---------------------------------------------------------------------------
// Row softmax: X [rows][C] -> att [rows][Cpad] (padding written as 0).
// ---------------------------------------------------------------------------
__global__ void softmax_rows(const float* __restrict__ X, float* __restrict__ att,
                             int C, int Cpad) {
    int row = blockIdx.x;
    int t = threadIdx.x;                 // 256 threads
    __shared__ float red[256];
    float x = (t < C) ? X[(size_t)row * C + t] : -3.402823466e38f;
    red[t] = x; __syncthreads();
    for (int s = 128; s > 0; s >>= 1) {
        if (t < s) red[t] = fmaxf(red[t], red[t + s]);
        __syncthreads();
    }
    float m = red[0]; __syncthreads();
    float e = (t < C) ? __expf(x - m) : 0.0f;
    red[t] = e; __syncthreads();
    for (int s = 128; s > 0; s >>= 1) {
        if (t < s) red[t] += red[t + s];
        __syncthreads();
    }
    float inv = 1.0f / red[0];
    if (t < Cpad) att[(size_t)row * Cpad + t] = e * inv;  // t>=C stores 0
}

// ---------------------------------------------------------------------------
// Per-class greedy IoU NMS (one block per class). Stable argsort(-score)
// by counting-rank (ties -> lower original index first, like jnp.argsort).
// Class 0 (background) always masked.
// ---------------------------------------------------------------------------
__global__ void nms_kernel(const float* __restrict__ att,     // [512][Cpad]
                           const float* __restrict__ boxes,   // [512][C][4]
                           float* __restrict__ mask,          // [512][Cpad]
                           int C, int Cpad) {
    int c = blockIdx.x;
    int t = threadIdx.x;                 // 256 threads
    __shared__ float sx1[N_NODES], sy1[N_NODES], sx2[N_NODES], sy2[N_NODES];
    __shared__ float sarea[N_NODES], ssc[N_NODES];
    __shared__ int order[N_NODES];
    __shared__ int keep[N_NODES];

    if (c == 0) {
        for (int i = t; i < N_NODES; i += blockDim.x)
            mask[(size_t)i * Cpad] = 0.0f;
        return;
    }

    for (int i = t; i < N_NODES; i += blockDim.x)
        ssc[i] = att[(size_t)i * Cpad + c];
    __syncthreads();

    for (int i = t; i < N_NODES; i += blockDim.x) {
        float si = ssc[i];
        int rnk = 0;
        for (int j = 0; j < N_NODES; ++j) {
            float sj = ssc[j];
            rnk += (sj > si) || (sj == si && j < i);
        }
        order[rnk] = i;
    }
    __syncthreads();

    for (int s = t; s < N_NODES; s += blockDim.x) {
        int i = order[s];
        const float* b = boxes + ((size_t)i * C + c) * 4;
        float x1 = b[0], y1 = b[1], x2 = b[2], y2 = b[3];
        sx1[s] = x1; sy1[s] = y1; sx2[s] = x2; sy2[s] = y2;
        sarea[s] = (x2 - x1) * (y2 - y1);
        keep[s] = 1;
    }
    __syncthreads();

    for (int i = 0; i < N_NODES; ++i) {
        if (keep[i]) {
            float x1 = sx1[i], y1 = sy1[i], x2 = sx2[i], y2 = sy2[i], ai = sarea[i];
            for (int j = i + 1 + t; j < N_NODES; j += blockDim.x) {
                float ix1 = fmaxf(x1, sx1[j]);
                float iy1 = fmaxf(y1, sy1[j]);
                float ix2 = fminf(x2, sx2[j]);
                float iy2 = fminf(y2, sy2[j]);
                float w = fmaxf(ix2 - ix1, 0.0f);
                float hh = fmaxf(iy2 - iy1, 0.0f);
                float inter = w * hh;
                float iou = inter / (ai + sarea[j] - inter + 1e-9f);
                if (iou > 0.3f) keep[j] = 0;
            }
        }
        __syncthreads();
    }

    for (int s = t; s < N_NODES; s += blockDim.x)
        mask[(size_t)order[s] * Cpad + c] = keep[s] ? 1.0f : 0.0f;
}

// ---------------------------------------------------------------------------
// attb = bf16(att * mask), zero in padding columns (k >= C).
// ---------------------------------------------------------------------------
__global__ void apply_mask_bf16(const float* __restrict__ att,
                                const float* __restrict__ mask,
                                __bf16* __restrict__ attb,
                                int C, int Cpad, int n) {
    int idx = blockIdx.x * blockDim.x + threadIdx.x;
    if (idx >= n) return;
    int k = idx % Cpad;
    float v = (k < C) ? att[idx] * mask[idx] : 0.0f;
    attb[idx] = f2bf(v);
}

// ---------------------------------------------------------------------------
// Launcher
// ---------------------------------------------------------------------------
extern "C" void kernel_launch(void* const* d_in, const int* in_sizes, int n_in,
                              void* d_out, int out_size, void* d_ws, size_t ws_size,
                              hipStream_t stream) {
    (void)in_sizes; (void)n_in; (void)out_size; (void)ws_size;

    const float* node_emb     = (const float*)d_in[0];   // [512, 2048]
    const float* edge_emb     = (const float*)d_in[1];   // [8192, 2048]
    const float* boxes        = (const float*)d_in[6];   // [512, 151, 4]
    const float* nodes_schema = (const float*)d_in[7];   // [2048, 151]
    const float* edges_schema = (const float*)d_in[8];   // [2048, 51]

    float* out      = (float*)d_out;
    float* raw_edge = out;                                  // 8192*51   = 417792
    float* h_edge   = out + 417792;                         // 8192*2048 = 16777216
    float* raw_node = out + 417792 + 16777216;              // 512*151   = 77312
    float* h_node   = raw_node + 77312;                     // 512*2048  = 1048576

    char* ws = (char*)d_ws;
    __bf16* bt_edge = (__bf16*)(ws + 0);        // [64][2048]  bf16 = 262144 B
    __bf16* bt_node = (__bf16*)(ws + 262144);   // [160][2048] bf16 = 655360 B
    __bf16* srows   = (__bf16*)(ws + 917504);   // [2048][160] bf16 = 655360 B
    float*  att     = (float*) (ws + 1572864);  // [512][160]  f32  = 327680 B
    float*  maskp   = (float*) (ws + 1900544);  // [512][160]  f32  = 327680 B
    __bf16* attb    = (__bf16*)(ws + 2228224);  // [512][160]  bf16 = 163840 B

    // 1) h_edge_emb = 0 (reference zeros it)
    zero_f4<<<16384, 256, 0, stream>>>((float4*)h_edge, 4194304);

    // 2) schema preps (bf16, transposed/padded for WMMA fragment striping)
    prep_bt<<<512, 256, 0, stream>>>(edges_schema, bt_edge, 2048, C_EDGE, 64);
    prep_bt<<<1280, 256, 0, stream>>>(nodes_schema, bt_node, 2048, C_NODE, 160);
    prep_rows<<<1280, 256, 0, stream>>>(nodes_schema, srows, 2048, C_NODE, 160);

    // 3) raw_edge_class = edge_emb @ edges_schema
    //    512 m-tiles, 4 waves/block, async-LDS B staging, 4 WMMAs per A frag.
    gemm_edge_wmma<<<128, 128, 0, stream>>>(edge_emb, bt_edge, raw_edge);

    // 4) raw_node_class = node_emb @ nodes_schema  (tiles: 32 x 10 = 320 waves)
    gemm_a32_bt16<<<80, 128, 0, stream>>>(node_emb, bt_node, raw_node,
                                          512, 2048, C_NODE, 10);

    // 5) att = softmax(raw_node_class, axis=1)
    softmax_rows<<<512, 256, 0, stream>>>(raw_node, att, C_NODE, 160);

    // 6) per-class NMS mask
    nms_kernel<<<151, 256, 0, stream>>>(att, boxes, maskp, C_NODE, 160);

    // 7) attb = bf16(att * mask)
    apply_mask_bf16<<<320, 256, 0, stream>>>(att, maskp, attb, C_NODE, 160,
                                             N_NODES * 160);

    // 8) h_node_emb = att @ nodes_schema.T  (tiles: 32 x 128 = 4096 waves)
    gemm_a16_b16<<<1024, 128, 0, stream>>>(attb, srows, h_node, 512, 160, 2048);
}
